// SemanticAwareAttention_81990925680867
// MI455X (gfx1250) — matrix operand order repaired and checked by reading, not compiled, not run on hardware
//
#include <hip/hip_runtime.h>
#include <hip/hip_bf16.h>

typedef __attribute__((ext_vector_type(16))) __bf16 v16bf;
typedef __attribute__((ext_vector_type(8)))  float  v8f;

#define B_   8
#define N_   16384
#define C_   40
#define CP_  64
#define E_   512
#define KD_  256

#define ROWS   64     // N-rows per block (4 strips of 16)
#define QP_ST  520    // bf16 stride for qp LDS  (1040 B, 16B multiple, bank-skewed)
#define ATT_ST 68     // f32 stride for attn LDS
#define CPB_ST 65     // f32 stride for coarse-logit LDS
#define AA_ST  72     // bf16 stride for attn-A LDS (144 B, 16B multiple)

// ---- fragment helpers -------------------------------------------------------
union FragU { uint4 u[2]; v16bf v; };

// packed B-fragment image in global: 32 lanes x 16 bf16 per 16x16(x32K) tile
__device__ __forceinline__ v16bf load_bfrag(const __bf16* tile_base, int lane) {
    FragU f;
    const uint4* p = (const uint4*)(tile_base + lane * 16);
    f.u[0] = p[0];
    f.u[1] = p[1];
    return f.v;
}

// A-fragment from row-major bf16 LDS row: per ISA 16-bit A layout,
// lanes 0-15 take K {k0..k0+7, k0+16..k0+23}; k0 already lane-adjusted.
__device__ __forceinline__ v16bf load_afrag(const __bf16* row, int k0) {
    FragU f;
    f.u[0] = *(const uint4*)(row + k0);
    f.u[1] = *(const uint4*)(row + k0 + 16);
    return f.v;
}

// A-fragment built from global f32 row (converts on the fly)
__device__ __forceinline__ v16bf build_afrag_f32(const float* row, int k0) {
    float4 a0 = *(const float4*)(row + k0);
    float4 a1 = *(const float4*)(row + k0 + 4);
    float4 a2 = *(const float4*)(row + k0 + 16);
    float4 a3 = *(const float4*)(row + k0 + 20);
    v16bf A;
    A[0]  = (__bf16)a0.x; A[1]  = (__bf16)a0.y; A[2]  = (__bf16)a0.z; A[3]  = (__bf16)a0.w;
    A[4]  = (__bf16)a1.x; A[5]  = (__bf16)a1.y; A[6]  = (__bf16)a1.z; A[7]  = (__bf16)a1.w;
    A[8]  = (__bf16)a2.x; A[9]  = (__bf16)a2.y; A[10] = (__bf16)a2.z; A[11] = (__bf16)a2.w;
    A[12] = (__bf16)a3.x; A[13] = (__bf16)a3.y; A[14] = (__bf16)a3.z; A[15] = (__bf16)a3.w;
    return A;
}

__device__ __forceinline__ v8f wmma_bf16(v16bf a, v16bf b, v8f c) {
    return __builtin_amdgcn_wmma_f32_16x16x32_bf16(false, a, false, b,
                                                   (short)0, c, false, false);
}

// ---- prep: pack Wq (E x E, row-major [f][e]) into B-fragment image ----------
__global__ void pack_wq_kernel(const float* __restrict__ Wq,
                               __bf16* __restrict__ WqP) {
    int idx = blockIdx.x * 256 + threadIdx.x;      // 512*512
    int f = idx >> 9;
    int e = idx & 511;
    float val = Wq[(size_t)f * E_ + e];
    int etile = e >> 5, ftile = f >> 4;
    int kl = e & 31, kk = (kl & 7) + ((kl >> 4) << 3);
    int lanei = (f & 15) + (((kl >> 3) & 1) ? 16 : 0);
    WqP[((size_t)(etile * 32 + ftile)) * 512 + lanei * 16 + kk] = (__bf16)val;
}

// ---- prep: kp = k*Wk^T + bk, vp = v*Wv^T + bv, packed as B-fragment images --
__global__ void pack_kv_kernel(const float* __restrict__ k,
                               const float* __restrict__ v,
                               const float* __restrict__ Wk,
                               const float* __restrict__ bk,
                               const float* __restrict__ Wv,
                               const float* __restrict__ bv,
                               __bf16* __restrict__ KpP,
                               __bf16* __restrict__ VpP) {
    int idx = blockIdx.x * 256 + threadIdx.x;      // 2 * 8 * 64 * 512 = 524288
    int which = idx >> 18;                         // 0 = kp, 1 = vp
    int r = idx & ((1 << 18) - 1);
    int b = r >> 15;
    int ce = r & 32767;
    int c = ce >> 9;                               // padded class 0..63
    int e = ce & 511;

    float val = 0.f;
    if (c < C_) {
        const float* src  = which ? v  : k;
        const float* W    = which ? Wv : Wk;
        const float* bias = which ? bv : bk;
        const float4* xr = (const float4*)(src + ((size_t)b * C_ + c) * KD_);
        const float4* wr = (const float4*)(W + (size_t)e * KD_);
        float acc = 0.f;
        for (int i = 0; i < KD_ / 4; ++i) {
            float4 a = xr[i], w = wr[i];
            acc += a.x * w.x + a.y * w.y + a.z * w.z + a.w * w.w;
        }
        val = acc + bias[e];
    }
    if (which == 0) {  // stage-2 B: K-dim = e (512), N-dim = c (64)
        int etile = e >> 5, ctile = c >> 4;
        int kl = e & 31, kk = (kl & 7) + ((kl >> 4) << 3);
        int lanei = (c & 15) + (((kl >> 3) & 1) ? 16 : 0);
        KpP[(size_t)b * 32768 + (size_t)(etile * 4 + ctile) * 512 + lanei * 16 + kk] =
            (__bf16)val;
    } else {           // stage-4 B: K-dim = c (64), N-dim = e (512)
        int ctile = c >> 5, ftile = e >> 4;
        int kl = c & 31, kk = (kl & 7) + ((kl >> 4) << 3);
        int lanei = (e & 15) + (((kl >> 3) & 1) ? 16 : 0);
        VpP[(size_t)b * 32768 + (size_t)(ctile * 32 + ftile) * 512 + lanei * 16 + kk] =
            (__bf16)val;
    }
}

// ---- fused: qp GEMM -> attn logits -> double softmax*prior -> out GEMM ------
__global__ __launch_bounds__(256)
void fused_attn_kernel(const float* __restrict__ q,
                       const float* __restrict__ coarse,
                       const float* __restrict__ bq,
                       const __bf16* __restrict__ WqP,
                       const __bf16* __restrict__ KpP,
                       const __bf16* __restrict__ VpP,
                       float* __restrict__ out) {
    __shared__ __align__(16) __bf16 qp[ROWS * QP_ST];    // 66560 B
    __shared__ __align__(16) float  att[ROWS * ATT_ST];  // 17408 B
    __shared__ __align__(16) float  cpb[C_ * CPB_ST];    // 10400 B
    __shared__ __align__(16) __bf16 aa[ROWS * AA_ST];    //  9216 B

    const int tid  = threadIdx.x;
    const int wave = tid >> 5;
    const int lane = tid & 31;
    const int lr   = lane & 15;          // row-in-tile (A/D) or col-in-tile (B/D)
    const int hi   = lane >> 4;          // lane-half select
    const int b    = blockIdx.x >> 8;    // N_/ROWS = 256 blocks per batch
    const int n0   = (blockIdx.x & 255) * ROWS;

    // zero attn-A padding; stage coarse logits (coalesced: tid%64 -> consecutive n)
    for (int i = tid; i < ROWS * AA_ST; i += 256) aa[i] = (__bf16)0.f;
    for (int i = tid; i < C_ * ROWS; i += 256) {
        int c = i >> 6, n = i & 63;
        cpb[c * CPB_ST + n] = coarse[((size_t)b * C_ + c) * N_ + n0 + n];
    }

    // ---------------- stage 1: qp = q * Wq^T + bq  (bf16 WMMA, K=512) --------
    // Each wave: 4 f-tiles (group = wave) x 4 M-strips; B loaded once per
    // (kt, ftile) and reused across the 4 strips -> 4x less Wq L2 traffic.
    {
        const int g = wave;                  // f-tile group 0..7
        const float* qbase = q + ((size_t)b * N_ + n0 + lr) * E_;
        v8f acc[4][4] = {};
        for (int kt = 0; kt < 16; ++kt) {
            const int k0 = kt * 32 + (hi ? 8 : 0);
            v16bf A[4];
            #pragma unroll
            for (int s = 0; s < 4; ++s)
                A[s] = build_afrag_f32(qbase + (size_t)s * 16 * E_, k0);
            #pragma unroll
            for (int j = 0; j < 4; ++j) {
                const int ft = g * 4 + j;
                v16bf Bf = load_bfrag(WqP + (size_t)(kt * 32 + ft) * 512, lane);
                #pragma unroll
                for (int s = 0; s < 4; ++s)
                    acc[s][j] = wmma_bf16(A[s], Bf, acc[s][j]);
            }
        }
        #pragma unroll
        for (int j = 0; j < 4; ++j) {
            const int col  = (g * 4 + j) * 16 + lr;
            const float bias = bq[col];
            #pragma unroll
            for (int s = 0; s < 4; ++s) {
                #pragma unroll
                for (int r = 0; r < 8; ++r) {
                    int row = s * 16 + r + 8 * hi;   // D layout: M = r + 8*(lane>=16)
                    qp[row * QP_ST + col] = (__bf16)(acc[s][j][r] + bias);
                }
            }
        }
    }
    __syncthreads();

    // ---------------- stage 2: attn = (qp * kp^T) * scale  (K=512) -----------
    // Each wave: 1 c-tile x 2 strips; B reused across the 2 strips.
    {
        const int ct = wave & 3;       // class tile 0..3 (C padded to 64)
        const int sp = wave >> 2;      // strip pair 0..1 -> strips {2sp, 2sp+1}
        v8f acc0 = {}, acc1 = {};
        const __bf16* qrow0 = qp + (sp * 32 + lr) * QP_ST;
        const __bf16* qrow1 = qp + (sp * 32 + 16 + lr) * QP_ST;
        const __bf16* kbase = KpP + (size_t)b * 32768 + (size_t)ct * 512 + (size_t)lane * 16;
        for (int kt = 0; kt < 16; ++kt) {
            const int k0 = kt * 32 + (hi ? 8 : 0);
            v16bf A0 = load_afrag(qrow0, k0);
            v16bf A1 = load_afrag(qrow1, k0);
            FragU f;
            const uint4* p = (const uint4*)(kbase + (size_t)kt * 2048);
            f.u[0] = p[0]; f.u[1] = p[1];
            acc0 = wmma_bf16(A0, f.v, acc0);
            acc1 = wmma_bf16(A1, f.v, acc1);
        }
        const float scale = 0.04419417382415922f;  // 1/sqrt(512)
        #pragma unroll
        for (int r = 0; r < 8; ++r) {
            int row0 = sp * 32 + r + 8 * hi;
            int row1 = row0 + 16;
            att[row0 * ATT_ST + ct * 16 + lr] = acc0[r] * scale;
            att[row1 * ATT_ST + ct * 16 + lr] = acc1[r] * scale;
        }
    }
    __syncthreads();

    // ---------------- stage 3: softmax(softmax(attn) * softmax(coarse^T)) ----
    if (tid < ROWS) {
        const int n = tid;
        float* arow = att + n * ATT_ST;
        float m1 = -1e30f;
        for (int c = 0; c < C_; ++c) m1 = fmaxf(m1, arow[c]);
        float s1 = 0.f;
        for (int c = 0; c < C_; ++c) { float e = __expf(arow[c] - m1); arow[c] = e; s1 += e; }
        float m2 = -1e30f;
        for (int c = 0; c < C_; ++c) m2 = fmaxf(m2, cpb[c * CPB_ST + n]);
        float s2 = 0.f;
        for (int c = 0; c < C_; ++c) s2 += __expf(cpb[c * CPB_ST + n] - m2);
        const float inv12 = 1.f / (s1 * s2);
        float m3 = -1e30f;
        for (int c = 0; c < C_; ++c) {
            float t = arow[c] * __expf(cpb[c * CPB_ST + n] - m2) * inv12;
            arow[c] = t;
            m3 = fmaxf(m3, t);
        }
        float s3 = 0.f;
        for (int c = 0; c < C_; ++c) { float e = __expf(arow[c] - m3); arow[c] = e; s3 += e; }
        const float inv3 = 1.f / s3;
        for (int c = 0; c < C_; ++c) aa[n * AA_ST + c] = (__bf16)(arow[c] * inv3);
    }
    __syncthreads();

    // ---------------- stage 4: out = attn * vp  (K = 64 padded) --------------
    // Each wave: 4 f-tiles x 4 strips; B reused across strips.
    {
        const int g = wave;
        v8f acc[4][4] = {};
        #pragma unroll
        for (int kt = 0; kt < 2; ++kt) {
            const int k0 = kt * 32 + (hi ? 8 : 0);
            v16bf A[4];
            #pragma unroll
            for (int s = 0; s < 4; ++s)
                A[s] = load_afrag(aa + (s * 16 + lr) * AA_ST, k0);
            #pragma unroll
            for (int j = 0; j < 4; ++j) {
                const int ft = g * 4 + j;
                v16bf Bf = load_bfrag(VpP + (size_t)b * 32768 + (size_t)(kt * 32 + ft) * 512,
                                      lane);
                #pragma unroll
                for (int s = 0; s < 4; ++s)
                    acc[s][j] = wmma_bf16(A[s], Bf, acc[s][j]);
            }
        }
        #pragma unroll
        for (int j = 0; j < 4; ++j) {
            const int col = (g * 4 + j) * 16 + lr;
            #pragma unroll
            for (int s = 0; s < 4; ++s) {
                #pragma unroll
                for (int r = 0; r < 8; ++r) {
                    int row = n0 + s * 16 + r + 8 * hi;
                    out[((size_t)b * N_ + row) * E_ + col] = acc[s][j][r];
                }
            }
        }
    }
}

// ---- launch -----------------------------------------------------------------
extern "C" void kernel_launch(void* const* d_in, const int* in_sizes, int n_in,
                              void* d_out, int out_size, void* d_ws, size_t ws_size,
                              hipStream_t stream) {
    const float* q      = (const float*)d_in[0];
    const float* k      = (const float*)d_in[1];
    const float* v      = (const float*)d_in[2];
    const float* coarse = (const float*)d_in[3];
    const float* Wq     = (const float*)d_in[4];
    const float* bq     = (const float*)d_in[5];
    const float* Wk     = (const float*)d_in[6];
    const float* bk     = (const float*)d_in[7];
    const float* Wv     = (const float*)d_in[8];
    const float* bv     = (const float*)d_in[9];
    float* out = (float*)d_out;

    // workspace: WqPack (512 KB) | KpPack (512 KB) | VpPack (512 KB)
    __bf16* WqP = reinterpret_cast<__bf16*>(d_ws);
    __bf16* KpP = WqP + 262144;
    __bf16* VpP = KpP + 262144;

    pack_wq_kernel<<<1024, 256, 0, stream>>>(Wq, WqP);
    pack_kv_kernel<<<2048, 256, 0, stream>>>(k, v, Wk, bk, Wv, bv, KpP, VpP);
    fused_attn_kernel<<<(B_ * N_) / ROWS, 256, 0, stream>>>(q, coarse, bq,
                                                            WqP, KpP, VpP, out);
}
